// TtBloomAttention_43843026157715
// MI455X (gfx1250) — matrix operand order repaired and checked by reading, not compiled, not run on hardware
//
#include <hip/hip_runtime.h>
#include <hip/hip_bf16.h>

// ---------------------------------------------------------------------------
// BLOOM attention block for MI455X (gfx1250), wave32, WMMA bf16 path.
// B=1, S=2048, H=4096, NH=32, HD=128.
// GEMMs: 128x128 block tile, 32x64 wave tile (8 WMMA / K-step / wave),
// double-buffered LDS staging via global_load_async_to_lds_b128 (ASYNCcnt);
// the in-order ASYNCcnt property lets s_wait_asynccnt 4 overlap the next
// stage's DMA with the current step's WMMAs.
// ---------------------------------------------------------------------------

typedef __bf16 bf16_t;
typedef __attribute__((ext_vector_type(16))) __bf16 v16bf;
typedef __attribute__((ext_vector_type(8)))  float  v8f;

#define S_LEN 2048
#define H_DIM 4096
#define NH_   32
#define HD_   128
#define K3H   12288   // 3*H
#define INV_NORM 0.08838834764831845f
#define MASK_MIN (-3.4028234663852886e38f)

// Build a v16bf fragment from two 16-byte chunks (j=0..7 from p0, j=8..15 from p1).
__device__ __forceinline__ v16bf load_frag16(const bf16_t* p0, const bf16_t* p1) {
    union { uint4 u[2]; v16bf v; } r;
    r.u[0] = *(const uint4*)p0;
    r.u[1] = *(const uint4*)p1;
    return r.v;
}

// LDS byte address = low 32 bits of the generic pointer (ISA: LDS aperture).
__device__ __forceinline__ uint32_t lds_addr32(const void* p) {
    return (uint32_t)(uintptr_t)p;
}

// Async 16-byte global -> LDS copy (per lane), tracked by ASYNCcnt.
__device__ __forceinline__ void async_ld_b128(uint32_t lds, const bf16_t* g) {
    asm volatile("global_load_async_to_lds_b128 %0, %1, off"
                 :: "v"(lds), "v"(g) : "memory");
}
__device__ __forceinline__ void wait_async0() {
    asm volatile("s_wait_asynccnt 0x0" ::: "memory");
}
// Wait until only the newest 4 async copies remain outstanding (older group done).
__device__ __forceinline__ void wait_async4() {
    asm volatile("s_wait_asynccnt 0x4" ::: "memory");
}

// ---------------------------------------------------------------------------
// fp32 -> bf16 conversion (grid-stride, plain layout)
// ---------------------------------------------------------------------------
__global__ void cvt_bf16_kernel(const float* __restrict__ src,
                                bf16_t* __restrict__ dst, long n) {
    long i = (long)blockIdx.x * blockDim.x + threadIdx.x;
    long stride = (long)gridDim.x * blockDim.x;
    for (; i < n; i += stride) dst[i] = (bf16_t)src[i];
}

// ---------------------------------------------------------------------------
// fp32 -> bf16 with transpose: dst[k][n] = src[n][k]. 32x32 LDS tiles.
// ---------------------------------------------------------------------------
__global__ __launch_bounds__(256) void cvt_transpose_bf16_kernel(
    const float* __restrict__ src, bf16_t* __restrict__ dst, int N, int K)
{
    __shared__ float tile[32][33];
    const int k0 = blockIdx.x * 32;
    const int n0 = blockIdx.y * 32;
    const int tx = threadIdx.x & 31;
    const int ty = threadIdx.x >> 5; // 0..7
    #pragma unroll
    for (int i = 0; i < 4; ++i) {
        const int n = ty + i * 8;
        tile[n][tx] = src[(size_t)(n0 + n) * K + k0 + tx];
    }
    __syncthreads();
    #pragma unroll
    for (int i = 0; i < 4; ++i) {
        const int k = ty + i * 8;
        dst[(size_t)(k0 + k) * N + n0 + tx] = (bf16_t)tile[tx][k];
    }
}

// ---------------------------------------------------------------------------
// Shared GEMM core: C[128x128] tile = A[M,K] @ WT[K,N] with bf16 WMMA.
// 8 waves, wave tile 32(M) x 64(N): acc[2][4]. K step 32, double-buffered LDS.
// ---------------------------------------------------------------------------
struct GemmAcc { v8f a[2][4]; };

#define A_TILE (128 * 32)
#define B_TILE (32 * 128)

template <int NW>
__device__ __forceinline__ GemmAcc gemm_core_128x128(
    const bf16_t* __restrict__ A, const bf16_t* __restrict__ WT,
    int m0, int n0, int Kdim, bf16_t* As, bf16_t* BsT)
{
    const int tid  = threadIdx.x;
    const int lane = tid & 31;
    const int wave = tid >> 5;
    const int rg = wave & 3;   // row group: rows rg*32 .. rg*32+31
    const int cg = wave >> 2;  // col group: cols cg*64 .. cg*64+63
    const int base0 = (lane < 16) ? 0 : 8;

    GemmAcc acc;
    #pragma unroll
    for (int r = 0; r < 2; ++r)
        #pragma unroll
        for (int t = 0; t < 4; ++t) acc.a[r][t] = (v8f){};

    // per-thread async copy slots: A tile = 512 x 16B chunks, BT tile = 512
    const bf16_t* gA[2]; const bf16_t* gB[2];
    uint32_t lA[2][2], lB[2][2];   // [buffer][slot]
    #pragma unroll
    for (int i = 0; i < 2; ++i) {
        const int c = tid + i * 256;
        gA[i] = A + (size_t)(m0 + (c >> 2)) * Kdim + (c & 3) * 8;
        gB[i] = WT + (size_t)(c >> 4) * NW + n0 + (c & 15) * 8;
        #pragma unroll
        for (int b = 0; b < 2; ++b) {
            lA[b][i] = lds_addr32(&As[b * A_TILE + (c >> 2) * 32 + (c & 3) * 8]);
            lB[b][i] = lds_addr32(&BsT[b * B_TILE + (c >> 4) * 128 + (c & 15) * 8]);
        }
    }

    // prologue: stage step 0 into buffer 0
    #pragma unroll
    for (int i = 0; i < 2; ++i) {
        async_ld_b128(lA[0][i], gA[i]);
        async_ld_b128(lB[0][i], gB[i]);
        gA[i] += 32;
        gB[i] += (size_t)32 * NW;
    }

    const int nsteps = Kdim / 32;
    for (int s = 0; s < nsteps; ++s) {
        const int buf = s & 1;
        if (s + 1 < nsteps) {
            const int nb = buf ^ 1;   // stage next step while computing this one
            #pragma unroll
            for (int i = 0; i < 2; ++i) {
                async_ld_b128(lA[nb][i], gA[i]);
                async_ld_b128(lB[nb][i], gB[i]);
                gA[i] += 32;
                gB[i] += (size_t)32 * NW;
            }
            __builtin_prefetch(gA[0], 0, 0);
            __builtin_prefetch(gB[0], 0, 0);
            wait_async4();   // older group of 4 done; newest 4 still in flight
        } else {
            wait_async0();
        }
        __syncthreads();

        const bf16_t* Asb  = As  + buf * A_TILE;
        const bf16_t* BsTb = BsT + buf * B_TILE;

        v16bf af[2];
        #pragma unroll
        for (int r = 0; r < 2; ++r) {
            const bf16_t* ap = &Asb[((rg * 2 + r) * 16 + (lane & 15)) * 32 + base0];
            af[r] = load_frag16(ap, ap + 16);
        }
        #pragma unroll
        for (int t = 0; t < 4; ++t) {
            const bf16_t* bp = &BsTb[cg * 64 + t * 16 + (lane & 15)];
            v16bf b;
            #pragma unroll
            for (int j = 0; j < 16; ++j) b[j] = bp[(base0 + (j & 7) + (j >> 3) * 16) * 128];
            #pragma unroll
            for (int r = 0; r < 2; ++r)
                acc.a[r][t] = __builtin_amdgcn_wmma_f32_16x16x32_bf16(
                    false, af[r], false, b, (short)0, acc.a[r][t], false, false);
        }
        __syncthreads();  // all waves done reading buf before it is re-staged
    }
    return acc;
}

// ---------------------------------------------------------------------------
// QKV GEMM: fused[S,3H] = hs @ Wqkv^T + b; WT given K-major [H][3H].
// Epilogue scatters into q [NH,S,HD], kT [NH,HD,S], v [NH,S,HD] (bf16).
// ---------------------------------------------------------------------------
__global__ __launch_bounds__(256) void qkv_gemm_kernel(
    const bf16_t* __restrict__ A, const bf16_t* __restrict__ WT,
    const float* __restrict__ bias,
    bf16_t* __restrict__ qb, bf16_t* __restrict__ kbT, bf16_t* __restrict__ vb)
{
    __shared__ bf16_t As[2 * A_TILE];
    __shared__ bf16_t BsT[2 * B_TILE];
    const int m0 = blockIdx.y * 128;
    const int n0 = blockIdx.x * 128;
    const int lane = threadIdx.x & 31;
    const int wave = threadIdx.x >> 5;
    const int rg = wave & 3, cg = wave >> 2;

    GemmAcc acc = gemm_core_128x128<K3H>(A, WT, m0, n0, H_DIM, As, BsT);

    #pragma unroll
    for (int r = 0; r < 2; ++r) {
        #pragma unroll
        for (int t = 0; t < 4; ++t) {
            const int cb = n0 + cg * 64 + t * 16;   // subtile base (16-aligned)
            const int head  = cb / 384;             // uniform within subtile
            const int which = (cb / 128) % 3;
            const int d     = (cb & 127) + (lane & 15);
            const int c     = cb + (lane & 15);
            const float bv = bias[c];
            if (which == 1) {  // K stored transposed: [head][d][s]
                #pragma unroll
                for (int i = 0; i < 8; ++i) {
                    const int m = m0 + (rg * 2 + r) * 16 + i + ((lane < 16) ? 0 : 8);
                    kbT[(size_t)head * HD_ * S_LEN + (size_t)d * S_LEN + m] =
                        (bf16_t)(acc.a[r][t][i] + bv);
                }
            } else {
                bf16_t* dst = (which == 0) ? qb : vb;
                #pragma unroll
                for (int i = 0; i < 8; ++i) {
                    const int m = m0 + (rg * 2 + r) * 16 + i + ((lane < 16) ? 0 : 8);
                    dst[(size_t)head * S_LEN * HD_ + (size_t)m * HD_ + d] =
                        (bf16_t)(acc.a[r][t][i] + bv);
                }
            }
        }
    }
}

// ---------------------------------------------------------------------------
// Dense GEMM: out[S,H] = ctx @ Wd^T + b_dense + residual (fp32 out).
// ---------------------------------------------------------------------------
__global__ __launch_bounds__(256) void dense_gemm_kernel(
    const bf16_t* __restrict__ A, const bf16_t* __restrict__ WT,
    const float* __restrict__ bias, const float* __restrict__ resid,
    float* __restrict__ out)
{
    __shared__ bf16_t As[2 * A_TILE];
    __shared__ bf16_t BsT[2 * B_TILE];
    const int m0 = blockIdx.y * 128;
    const int n0 = blockIdx.x * 128;
    const int lane = threadIdx.x & 31;
    const int wave = threadIdx.x >> 5;
    const int rg = wave & 3, cg = wave >> 2;

    GemmAcc acc = gemm_core_128x128<H_DIM>(A, WT, m0, n0, H_DIM, As, BsT);

    #pragma unroll
    for (int r = 0; r < 2; ++r) {
        #pragma unroll
        for (int t = 0; t < 4; ++t) {
            const int c = n0 + cg * 64 + t * 16 + (lane & 15);
            const float bv = bias[c];
            #pragma unroll
            for (int i = 0; i < 8; ++i) {
                const int m = m0 + (rg * 2 + r) * 16 + i + ((lane < 16) ? 0 : 8);
                out[(size_t)m * H_DIM + c] =
                    acc.a[r][t][i] + bv + resid[(size_t)m * H_DIM + c];
            }
        }
    }
}

// ---------------------------------------------------------------------------
// Attention: one block per (head, 16-query tile). 256 threads (8 waves).
// LDS: fp32 scores [16][2048] + bf16 probs [16][2048] + scratch.
// ---------------------------------------------------------------------------
__global__ __launch_bounds__(256) void attn_kernel(
    const bf16_t* __restrict__ qb, const bf16_t* __restrict__ kbT,
    const bf16_t* __restrict__ vb, const float* __restrict__ alibi,
    bf16_t* __restrict__ ctxb)
{
    extern __shared__ char smem[];
    float*  sc  = (float*)smem;                        // [16][2048] fp32
    bf16_t* pb  = (bf16_t*)(smem + 16 * 2048 * 4);     // [16][2048] bf16
    float*  red = (float*)(smem + 16 * 2048 * 6);      // [16][16]

    const int tid  = threadIdx.x;
    const int lane = tid & 31;
    const int wave = tid >> 5;
    const int head = blockIdx.x >> 7;
    const int qt0  = (blockIdx.x & 127) * 16;
    const int base0 = (lane < 16) ? 0 : 8;

    const bf16_t* Q  = qb  + (size_t)head * S_LEN * HD_;
    const bf16_t* KT = kbT + (size_t)head * HD_ * S_LEN;   // [d][s]
    const bf16_t* V  = vb  + (size_t)head * S_LEN * HD_;

    v16bf qf[4];
    {
        const bf16_t* qp = Q + (size_t)(qt0 + (lane & 15)) * HD_;
        #pragma unroll
        for (int s = 0; s < 4; ++s)
            qf[s] = load_frag16(qp + s * 32 + base0, qp + s * 32 + base0 + 16);
    }

    // ---- phase 1: scores = inv_norm * Q K^T + alibi, causal mask ----
    for (int nt = wave; nt < 128; nt += 8) {
        v8f acc = {};
        #pragma unroll
        for (int s = 0; s < 4; ++s) {
            const bf16_t* bp = KT + (size_t)(s * 32) * S_LEN + nt * 16 + (lane & 15);
            v16bf b;
            #pragma unroll
            for (int j = 0; j < 16; ++j) b[j] = bp[(size_t)(base0 + (j & 7) + (j >> 3) * 16) * S_LEN];
            acc = __builtin_amdgcn_wmma_f32_16x16x32_bf16(
                false, qf[s], false, b, (short)0, acc, false, false);
        }
        const int key = nt * 16 + (lane & 15);
        const float al = alibi[(size_t)head * S_LEN + key];
        #pragma unroll
        for (int i = 0; i < 8; ++i) {
            const int r = i + ((lane < 16) ? 0 : 8);
            float v = acc[i] * INV_NORM + al;
            if (key > qt0 + r) v = MASK_MIN;
            sc[r * 2048 + key] = v;
        }
    }
    __syncthreads();

    // ---- phase 2: row softmax (16 threads per row) ----
    {
        const int r = tid >> 4, c0 = tid & 15;
        float lm = MASK_MIN;
        for (int c = c0; c < 2048; c += 16) lm = fmaxf(lm, sc[r * 2048 + c]);
        red[r * 16 + c0] = lm;
        __syncthreads();
        if (c0 == 0) {
            float mm = red[r * 16];
            for (int j = 1; j < 16; ++j) mm = fmaxf(mm, red[r * 16 + j]);
            red[r * 16] = mm;
        }
        __syncthreads();
        const float rowmax = red[r * 16];
        __syncthreads();
        float ls = 0.f;
        for (int c = c0; c < 2048; c += 16) {
            const float e = __expf(sc[r * 2048 + c] - rowmax);
            sc[r * 2048 + c] = e;
            ls += e;
        }
        red[r * 16 + c0] = ls;
        __syncthreads();
        if (c0 == 0) {
            float ss = red[r * 16];
            for (int j = 1; j < 16; ++j) ss += red[r * 16 + j];
            red[r * 16] = ss;
        }
        __syncthreads();
        const float inv = 1.0f / red[r * 16];
        for (int c = c0; c < 2048; c += 16)
            pb[r * 2048 + c] = (bf16_t)(sc[r * 2048 + c] * inv);
    }
    __syncthreads();

    // ---- phase 3: ctx = P @ V ; wave w owns HD columns [w*16, w*16+16) ----
    {
        const int dt = wave;
        v8f acc = {};
        const bf16_t* pp = &pb[(lane & 15) * 2048];
        for (int k0 = 0; k0 < 2048; k0 += 32) {
            const v16bf a = load_frag16(pp + k0 + base0, pp + k0 + base0 + 16);
            const bf16_t* vp = &V[(size_t)k0 * HD_ + dt * 16 + (lane & 15)];
            v16bf b;
            #pragma unroll
            for (int j = 0; j < 16; ++j) b[j] = vp[(base0 + (j & 7) + (j >> 3) * 16) * HD_];
            acc = __builtin_amdgcn_wmma_f32_16x16x32_bf16(
                false, a, false, b, (short)0, acc, false, false);
        }
        const int d = dt * 16 + (lane & 15);
        #pragma unroll
        for (int i = 0; i < 8; ++i) {
            const int m = qt0 + i + ((lane < 16) ? 0 : 8);
            ctxb[(size_t)m * H_DIM + head * HD_ + d] = (bf16_t)acc[i];
        }
    }
}

// ---------------------------------------------------------------------------
// Launch
// ---------------------------------------------------------------------------
extern "C" void kernel_launch(void* const* d_in, const int* in_sizes, int n_in,
                              void* d_out, int out_size, void* d_ws, size_t ws_size,
                              hipStream_t stream) {
    const float* hs    = (const float*)d_in[0];
    const float* resid = (const float*)d_in[1];
    const float* alibi = (const float*)d_in[2];
    // d_in[3] attention_mask: causal mask computed analytically
    const float* Wqkv  = (const float*)d_in[4];
    const float* bqkv  = (const float*)d_in[5];
    const float* Wd    = (const float*)d_in[6];
    const float* bd    = (const float*)d_in[7];
    float* out = (float*)d_out;

    char* ws = (char*)d_ws;
    bf16_t* hsb    = (bf16_t*)ws;  ws += (size_t)S_LEN * H_DIM * 2;
    bf16_t* wqkvT  = (bf16_t*)ws;  ws += (size_t)K3H * H_DIM * 2;   // [H][3H]
    bf16_t* wdT    = (bf16_t*)ws;  ws += (size_t)H_DIM * H_DIM * 2; // [H][H]
    bf16_t* qb     = (bf16_t*)ws;  ws += (size_t)NH_ * S_LEN * HD_ * 2;
    bf16_t* kbT    = (bf16_t*)ws;  ws += (size_t)NH_ * S_LEN * HD_ * 2; // [NH][HD][S]
    bf16_t* vb     = (bf16_t*)ws;  ws += (size_t)NH_ * S_LEN * HD_ * 2;
    bf16_t* ctxb   = (bf16_t*)ws;  ws += (size_t)S_LEN * H_DIM * 2;

    // 1) fp32 -> bf16 conversions (weights transposed to K-major)
    cvt_bf16_kernel<<<2048, 256, 0, stream>>>(hs, hsb, (long)S_LEN * H_DIM);
    cvt_transpose_bf16_kernel<<<dim3(H_DIM / 32, K3H / 32), 256, 0, stream>>>(
        Wqkv, wqkvT, K3H, H_DIM);
    cvt_transpose_bf16_kernel<<<dim3(H_DIM / 32, H_DIM / 32), 256, 0, stream>>>(
        Wd, wdT, H_DIM, H_DIM);

    // 2) QKV projection (128x128 tiles)
    qkv_gemm_kernel<<<dim3(K3H / 128, S_LEN / 128), 256, 0, stream>>>(
        hsb, wqkvT, bqkv, qb, kbT, vb);

    // 3) attention (one block per head x 16-query tile)
    const int attn_smem = 16 * 2048 * 4 + 16 * 2048 * 2 + 16 * 16 * 4; // 197632 B
    hipFuncSetAttribute((const void*)attn_kernel,
                        hipFuncAttributeMaxDynamicSharedMemorySize, attn_smem);
    attn_kernel<<<NH_ * (S_LEN / 16), 256, attn_smem, stream>>>(
        qb, kbT, vb, alibi, ctxb);

    // 4) dense projection + bias + residual
    dense_gemm_kernel<<<dim3(H_DIM / 128, S_LEN / 128), 256, 0, stream>>>(
        ctxb, wdT, bd, resid, out);
}
